// HeatRegNet_29205777613587
// MI455X (gfx1250) — matrix operand order repaired and checked by reading, not compiled, not run on hardware
//
#include <hip/hip_runtime.h>
#include <hip/hip_bf16.h>

typedef __attribute__((ext_vector_type(16))) __bf16 v16bf;
typedef __attribute__((ext_vector_type(8)))  float  v8f;

#define NPTS 4096
#define KNN  16
#define EPSV 1e-5f

__device__ __forceinline__ unsigned short f2bf(float f) {
  unsigned u = __float_as_uint(f);
  u = (u + 0x7FFFu + ((u >> 16) & 1u)) >> 16;
  return (unsigned short)u;
}
__device__ __forceinline__ float bf2f(unsigned short h) {
  return __uint_as_float(((unsigned)h) << 16);
}

__global__ void k_zero(float* p, int n) {
  int i = blockIdx.x * blockDim.x + threadIdx.x;
  if (i < n) p[i] = 0.f;
}

// Pointwise conv (o,c) + optional fused GN(prev stats)+ReLU on input; emits raw y
// and accumulates this layer's group sums. Layout: point-major [B][N][C].
__global__ void k_conv_gn(const float* __restrict__ xin, const float* __restrict__ W,
                          const float* __restrict__ bias,
                          const float* __restrict__ pStats, const float* __restrict__ pGamma,
                          const float* __restrict__ pBeta,
                          float* __restrict__ yout, float* __restrict__ oStats,
                          int Cin, int Cout) {
  __shared__ float sW[8192];
  __shared__ float sSc[32], sSh[32];
  __shared__ float sRed[8];
  int b = blockIdx.y;
  int tid = threadIdx.x;
  int nW = Cout * Cin;
  for (int i = tid; i < nW; i += blockDim.x) sW[i] = W[i];
  if (tid < 8) sRed[tid] = 0.f;
  if (tid < Cin) {
    if (pStats) {
      int gpc = Cin >> 2;
      int g = tid / gpc;
      float cnt = (float)gpc * 4096.0f;
      float mu  = pStats[(b * 4 + g) * 2] / cnt;
      float var = pStats[(b * 4 + g) * 2 + 1] / cnt - mu * mu;
      float sc  = pGamma[tid] * rsqrtf(var + EPSV);
      sSc[tid] = sc; sSh[tid] = pBeta[tid] - mu * sc;
    } else { sSc[tid] = 1.f; sSh[tid] = 0.f; }
  }
  __syncthreads();
  int n = blockIdx.x * blockDim.x + tid;
  float x[32];
  const float* xrow = xin + (size_t)(b * NPTS + n) * Cin;
  bool relu = (pStats != nullptr);
  for (int c = 0; c < Cin; c++) {
    float v = xrow[c] * sSc[c] + sSh[c];
    x[c] = relu ? fmaxf(v, 0.f) : v;
  }
  float gs[4] = {0, 0, 0, 0}, gss[4] = {0, 0, 0, 0};
  int gpc_o = Cout >> 2;
  float* yrow = yout + (size_t)(b * NPTS + n) * Cout;
  for (int o = 0; o < Cout; o++) {
    float acc = bias[o];
    const float* wr = &sW[o * Cin];
    for (int c = 0; c < Cin; c++) acc += wr[c] * x[c];
    yrow[o] = acc;
    int g = o / gpc_o;
    gs[g] += acc; gss[g] += acc * acc;
  }
  for (int g = 0; g < 4; g++) {
    atomicAdd(&sRed[g * 2], gs[g]);
    atomicAdd(&sRed[g * 2 + 1], gss[g]);
  }
  __syncthreads();
  if (tid < 8) atomicAdd(&oStats[b * 8 + tid], sRed[tid]);
}

// GN+ReLU on last global-feat layer, max over N -> gfeat[b][256]
__global__ void k_gn_max(const float* __restrict__ y4, const float* __restrict__ st,
                         const float* __restrict__ gamma, const float* __restrict__ beta,
                         float* __restrict__ gfeat) {
  __shared__ float sm[128];
  int o = blockIdx.x, b = blockIdx.y, tid = threadIdx.x;
  int g = o >> 6;
  float cnt = 64.0f * 4096.0f;
  float mu  = st[(b * 4 + g) * 2] / cnt;
  float var = st[(b * 4 + g) * 2 + 1] / cnt - mu * mu;
  float sc  = gamma[o] * rsqrtf(var + EPSV);
  float sh  = beta[o] - mu * sc;
  float m = 0.f;  // post-ReLU values are >= 0
  for (int n = tid; n < NPTS; n += 128) {
    float v = fmaxf(y4[(size_t)(b * NPTS + n) * 256 + o] * sc + sh, 0.f);
    m = fmaxf(m, v);
  }
  sm[tid] = m; __syncthreads();
  for (int s = 64; s > 0; s >>= 1) {
    if (tid < s) sm[tid] = fmaxf(sm[tid], sm[tid + s]);
    __syncthreads();
  }
  if (tid == 0) gfeat[b * 256 + o] = sm[0];
}

// Brute-force 16-NN (unordered set == softmax-sum invariant), emits cand = km[idx]-kf
__global__ void k_knn(const float* __restrict__ kf, const float* __restrict__ km,
                      float* __restrict__ cand) {
  __shared__ float sx[256], sy[256], sz[256];
  int b = blockIdx.y;
  int n = blockIdx.x * 256 + threadIdx.x;
  const float* f = kf + (size_t)(b * NPTS + n) * 3;
  float fx = f[0], fy = f[1], fz = f[2];
  float bd[KNN]; int bi[KNN];
  for (int k = 0; k < KNN; k++) { bd[k] = 3.4e38f; bi[k] = 0; }
  float wd = 3.4e38f; int wi = 0;
  for (int c0 = 0; c0 < NPTS; c0 += 256) {
    __syncthreads();
    const float* mp = km + (size_t)(b * NPTS + c0 + threadIdx.x) * 3;
    sx[threadIdx.x] = mp[0]; sy[threadIdx.x] = mp[1]; sz[threadIdx.x] = mp[2];
    __syncthreads();
    for (int j = 0; j < 256; j++) {
      float dx = sx[j] - fx, dy = sy[j] - fy, dz = sz[j] - fz;
      float d = dx * dx + dy * dy + dz * dz;
      if (d < wd) {
        bd[wi] = d; bi[wi] = c0 + j;
        wd = bd[0]; wi = 0;
        for (int t = 1; t < KNN; t++) { if (bd[t] > wd) { wd = bd[t]; wi = t; } }
      }
    }
  }
  float* cr = cand + (size_t)((b * NPTS + n) * KNN) * 3;
  for (int k = 0; k < KNN; k++) {
    const float* mp = km + (size_t)(b * NPTS + bi[k]) * 3;
    cr[k * 3 + 0] = mp[0] - fx; cr[k * 3 + 1] = mp[1] - fy; cr[k * 3 + 2] = mp[2] - fz;
  }
}

// base0[b][o] = db0[o] + W0[o,6:262]@gfix + W0[o,262:518]@gmov (broadcast channels)
__global__ void k_base0(const float* __restrict__ dw0, const float* __restrict__ db0,
                        const float* __restrict__ gfix, const float* __restrict__ gmov,
                        float* __restrict__ base0) {
  int b = blockIdx.x, o = threadIdx.x;
  const float* wr = dw0 + (size_t)o * 518;
  float acc = db0[o];
  for (int c = 0; c < 256; c++)
    acc += wr[6 + c] * gfix[b * 256 + c] + wr[262 + c] * gmov[b * 256 + c];
  base0[b * 256 + o] = acc;
}

// Pass over all positions: compute y0 (256x6 reduced form), accumulate GN stats0
__global__ void k_stats0(const float* __restrict__ kf, const float* __restrict__ cand,
                         const float* __restrict__ base0, const float* __restrict__ dw0,
                         float* __restrict__ stats0) {
  __shared__ float sRed[8];
  int b = blockIdx.y, tid = threadIdx.x;
  if (tid < 8) sRed[tid] = 0.f;
  __syncthreads();
  int p = blockIdx.x * 256 + tid;
  int n = p >> 4, k = p & 15;
  const float* f = kf + (size_t)(b * NPTS + n) * 3;
  const float* c = cand + (size_t)((b * NPTS + n) * KNN + k) * 3;
  float in6[6] = {f[0], f[1], f[2], c[0], c[1], c[2]};
  for (int g = 0; g < 4; g++) {
    float s = 0.f, ss = 0.f;
    for (int o = g * 64; o < g * 64 + 64; o++) {
      const float* wr = dw0 + (size_t)o * 518;
      float y = base0[b * 256 + o];
      for (int j = 0; j < 6; j++) y += wr[j] * in6[j];
      s += y; ss += y * y;
    }
    atomicAdd(&sRed[g * 2], s); atomicAdd(&sRed[g * 2 + 1], ss);
  }
  __syncthreads();
  if (tid < 8) atomicAdd(&stats0[b * 8 + tid], sRed[tid]);
}

// Swizzle W1 (128x256 f32) into bf16 WMMA A-matrix fragments:
// frag element j of lane L in (rt,q): M=rt*16+(L&15), K=q*32+((j>>3)*16+(L>>4)*8+(j&7))
__global__ void k_w1prep(const float* __restrict__ dw1, unsigned short* __restrict__ w1bf) {
  int e0 = blockIdx.x * 256 + threadIdx.x;
  for (int i = 0; i < 4; i++) {
    int e = e0 + i * 8192;
    int j = e & 15;
    int lane = (e >> 4) & 31;
    int q = (e >> 9) & 7;
    int rt = e >> 12;
    int M = rt * 16 + (lane & 15);
    int K = q * 32 + ((j >> 3) * 16 + ((lane >> 4) * 8) + (j & 7));
    w1bf[e] = f2bf(dw1[M * 256 + K]);
  }
}

// Fused: recompute layer0 (reduced 256x6 form) -> GN0+ReLU -> bf16 LDS tile in
// WMMA B-layout -> 8 waves x 8 chained v_wmma_f32_16x16x32_bf16 (K=256) -> +bias
// -> GN1 stats (shfl+atomic) -> y1 bf16 store. One block per (b, fixed point n).
__global__ void k_layer01(const float* __restrict__ kf, const float* __restrict__ cand,
                          const float* __restrict__ base0, const float* __restrict__ dw0,
                          const float* __restrict__ stats0,
                          const float* __restrict__ dg0, const float* __restrict__ dbt0,
                          const unsigned short* __restrict__ w1bf,
                          const float* __restrict__ db1,
                          unsigned short* __restrict__ y1, float* __restrict__ stats1) {
  __shared__ float sc0[256], sh0[256];
  __shared__ float skf[3], scand[48];
  __shared__ __align__(32) unsigned short xtile[8 * 512];
  int b = blockIdx.y;
  int n = blockIdx.x;
  int tid = threadIdx.x;
  {
    int o = tid, g = o >> 6;
    float cnt = 64.0f * 65536.0f;
    float mu  = stats0[b * 8 + g * 2] / cnt;
    float var = stats0[b * 8 + g * 2 + 1] / cnt - mu * mu;
    float sc  = dg0[o] * rsqrtf(var + EPSV);
    sc0[o] = sc; sh0[o] = dbt0[o] - mu * sc;
  }
  if (tid < 3)  skf[tid]   = kf[(size_t)(b * NPTS + n) * 3 + tid];
  if (tid < 48) scand[tid] = cand[(size_t)((b * NPTS + n) * KNN) * 3 + tid];
  __syncthreads();
  {  // build 256x16 activation tile in B-matrix layout (bf16)
    int k = tid & 15, oc = tid >> 4;
    int q = oc >> 1, hl = oc & 1;
    unsigned short* dst = &xtile[q * 512 + (hl * 16 + k) * 16];
#pragma unroll
    for (int jj = 0; jj < 16; jj++) {
      int o = q * 32 + (jj >> 3) * 16 + hl * 8 + (jj & 7);
      const float* wr = dw0 + (size_t)o * 518;
      float y = base0[b * 256 + o]
              + wr[0] * skf[0] + wr[1] * skf[1] + wr[2] * skf[2]
              + wr[3] * scand[k * 3 + 0] + wr[4] * scand[k * 3 + 1] + wr[5] * scand[k * 3 + 2];
      y = fmaxf(y * sc0[o] + sh0[o], 0.f);
      dst[jj] = f2bf(y);
    }
  }
  __syncthreads();
  int lane = tid & 31;
  int rt = tid >> 5;  // output row-tile 0..7 (128 channels / 16)
  v8f acc = {0.f, 0.f, 0.f, 0.f, 0.f, 0.f, 0.f, 0.f};
#pragma unroll
  for (int q = 0; q < 8; q++) {
    v16bf a  = *(const v16bf*)(w1bf + (size_t)((rt * 8 + q) * 32 + lane) * 16);
    v16bf bm = *(const v16bf*)(&xtile[q * 512 + lane * 16]);
    acc = __builtin_amdgcn_wmma_f32_16x16x32_bf16(false, a, false, bm,
                                                  (short)0, acc, false, false);
  }
  int hl = lane >> 4, col = lane & 15;
  int Mbase = rt * 16 + hl * 8;
  float s = 0.f, ss = 0.f;
  unsigned short h[8];
#pragma unroll
  for (int v = 0; v < 8; v++) {
    float y = acc[v] + db1[Mbase + v];
    s += y; ss += y * y;
    h[v] = f2bf(y);
  }
  unsigned short* drow = y1 + (size_t)(b * 65536 + n * 16 + col) * 128 + Mbase;
  uint4 pk;
  pk.x = (unsigned)h[0] | ((unsigned)h[1] << 16);
  pk.y = (unsigned)h[2] | ((unsigned)h[3] << 16);
  pk.z = (unsigned)h[4] | ((unsigned)h[5] << 16);
  pk.w = (unsigned)h[6] | ((unsigned)h[7] << 16);
  *(uint4*)drow = pk;
  for (int i = 16; i >= 1; i >>= 1) {
    s  += __shfl_xor(s, i, 32);
    ss += __shfl_xor(ss, i, 32);
  }
  if (lane == 0) {
    int g = rt >> 1;  // 128ch/4 groups: all 256 tile values share one group
    atomicAdd(&stats1[b * 8 + g * 2], s);
    atomicAdd(&stats1[b * 8 + g * 2 + 1], ss);
  }
}

// GN1+ReLU -> layer2 (128-dot) -> softmax over K=16 -> weighted cand sum
__global__ void k_final(const unsigned short* __restrict__ y1, const float* __restrict__ stats1,
                        const float* __restrict__ dg1, const float* __restrict__ dbt1,
                        const float* __restrict__ dw2, const float* __restrict__ db2,
                        const float* __restrict__ cand, float* __restrict__ out) {
  __shared__ float sc1[128], sh1[128], sw2[128];
  int b = blockIdx.y, tid = threadIdx.x;
  for (int c = tid; c < 128; c += 64) {
    int g = c >> 5;
    float cnt = 32.0f * 65536.0f;
    float mu  = stats1[b * 8 + g * 2] / cnt;
    float var = stats1[b * 8 + g * 2 + 1] / cnt - mu * mu;
    float sc  = dg1[c] * rsqrtf(var + EPSV);
    sc1[c] = sc; sh1[c] = dbt1[c] - mu * sc;
    sw2[c] = dw2[c];
  }
  __syncthreads();
  int n = blockIdx.x * 64 + tid;
  float logit[16];
  float bias2 = db2[0];
  for (int k = 0; k < 16; k++) {
    const unsigned* row = (const unsigned*)(y1 + (size_t)(b * 65536 + n * 16 + k) * 128);
    float acc = bias2;
    for (int c2 = 0; c2 < 64; c2++) {
      unsigned w = row[c2];
      int c = c2 * 2;
      float f0 = bf2f((unsigned short)(w & 0xFFFFu));
      float f1 = bf2f((unsigned short)(w >> 16));
      acc += fmaxf(f0 * sc1[c] + sh1[c], 0.f) * sw2[c];
      acc += fmaxf(f1 * sc1[c + 1] + sh1[c + 1], 0.f) * sw2[c + 1];
    }
    logit[k] = acc;
  }
  float m = logit[0];
  for (int k = 1; k < 16; k++) m = fmaxf(m, logit[k]);
  float ssum = 0.f;
  for (int k = 0; k < 16; k++) { logit[k] = __expf(logit[k] - m); ssum += logit[k]; }
  float inv = 1.f / ssum;
  float ox = 0.f, oy = 0.f, oz = 0.f;
  const float* cr = cand + (size_t)((b * NPTS + n) * KNN) * 3;
  for (int k = 0; k < 16; k++) {
    float w = logit[k] * inv;
    ox += w * cr[k * 3 + 0]; oy += w * cr[k * 3 + 1]; oz += w * cr[k * 3 + 2];
  }
  float* orow = out + (size_t)(b * NPTS + n) * 3;
  orow[0] = ox; orow[1] = oy; orow[2] = oz;
}

extern "C" void kernel_launch(void* const* d_in, const int* in_sizes, int n_in,
                              void* d_out, int out_size, void* d_ws, size_t ws_size,
                              hipStream_t stream) {
  (void)in_sizes; (void)n_in; (void)out_size; (void)ws_size;
  const float* kf = (const float*)d_in[0];
  const float* km = (const float*)d_in[1];
  const float *gw[5], *gb[5], *gg[5], *gbt[5];
  for (int i = 0; i < 5; i++) {
    gw[i]  = (const float*)d_in[2 + 4 * i];
    gb[i]  = (const float*)d_in[3 + 4 * i];
    gg[i]  = (const float*)d_in[4 + 4 * i];
    gbt[i] = (const float*)d_in[5 + 4 * i];
  }
  const float* dw0 = (const float*)d_in[22], *db0 = (const float*)d_in[23];
  const float* dg0 = (const float*)d_in[24], *dbt0 = (const float*)d_in[25];
  const float* dw1 = (const float*)d_in[26], *db1 = (const float*)d_in[27];
  const float* dg1 = (const float*)d_in[28], *dbt1 = (const float*)d_in[29];
  const float* dw2 = (const float*)d_in[30], *db2 = (const float*)d_in[31];

  char* w = (char*)d_ws;
  float* gstatsF = (float*)(w + 0);      // [5][B][4][2] = 80 f32
  float* gstatsM = (float*)(w + 512);
  float* stats0  = (float*)(w + 1024);   // [B][4][2]
  float* stats1  = (float*)(w + 1152);
  float* gfeatF  = (float*)(w + 2048);   // [B][256]
  float* gfeatM  = (float*)(w + 4096);
  float* base0   = (float*)(w + 6144);   // [B][256]
  float* cand    = (float*)(w + 8192);   // [B][N][16][3] f32 = 1.5MB
  float* act0    = (float*)(w + 8192 + 1572864);          // [B][N][32]
  float* act1    = act0 + 2 * NPTS * 32;
  float* y4      = act1 + 2 * NPTS * 32;                  // [B][N][256]
  unsigned short* w1bf = (unsigned short*)(y4 + 2 * NPTS * 256);  // 32768 bf16
  unsigned short* y1   = w1bf + 32768;                    // [B][65536][128] bf16

  k_zero<<<1, 512, 0, stream>>>((float*)w, 512);

  for (int cl = 0; cl < 2; cl++) {
    const float* pts = cl ? km : kf;
    float* gst = cl ? gstatsM : gstatsF;
    float* gfe = cl ? gfeatM : gfeatF;
    dim3 gr(32, 2);
    k_conv_gn<<<gr, 128, 0, stream>>>(pts,  gw[0], gb[0], nullptr, nullptr, nullptr, act0, gst + 0,  3, 16);
    k_conv_gn<<<gr, 128, 0, stream>>>(act0, gw[1], gb[1], gst + 0,  gg[0], gbt[0],   act1, gst + 16, 16, 16);
    k_conv_gn<<<gr, 128, 0, stream>>>(act1, gw[2], gb[2], gst + 16, gg[1], gbt[1],   act0, gst + 32, 16, 16);
    k_conv_gn<<<gr, 128, 0, stream>>>(act0, gw[3], gb[3], gst + 32, gg[2], gbt[2],   act1, gst + 48, 16, 32);
    k_conv_gn<<<gr, 128, 0, stream>>>(act1, gw[4], gb[4], gst + 48, gg[3], gbt[3],   y4,   gst + 64, 32, 256);
    k_gn_max<<<dim3(256, 2), 128, 0, stream>>>(y4, gst + 64, gg[4], gbt[4], gfe);
  }

  k_knn<<<dim3(16, 2), 256, 0, stream>>>(kf, km, cand);
  k_base0<<<2, 256, 0, stream>>>(dw0, db0, gfeatF, gfeatM, base0);
  k_stats0<<<dim3(256, 2), 256, 0, stream>>>(kf, cand, base0, dw0, stats0);
  k_w1prep<<<32, 256, 0, stream>>>(dw1, w1bf);
  k_layer01<<<dim3(4096, 2), 256, 0, stream>>>(kf, cand, base0, dw0, stats0,
                                               dg0, dbt0, w1bf, db1, y1, stats1);
  k_final<<<dim3(64, 2), 64, 0, stream>>>(y1, stats1, dg1, dbt1, dw2, db2, cand,
                                          (float*)d_out);
}
